// HeteroClassifier_25211458027582
// MI455X (gfx1250) — matrix operand order repaired and checked by reading, not compiled, not run on hardware
//
#include <hip/hip_runtime.h>

#define Nn 100000
#define Bb 64
#define Rr 3
#define Ee 200000
#define Dd 128
#define Cc 10

typedef __attribute__((ext_vector_type(16))) __bf16 bf16x16;
typedef __attribute__((ext_vector_type(8)))  __bf16 bf16x8;
typedef __attribute__((ext_vector_type(4)))  __bf16 bf16x4;
typedef __attribute__((ext_vector_type(8)))  float  f32x8;

// ---------------- utility kernels ----------------

__global__ void k_zero(float* __restrict__ p, int n) {
    int i = blockIdx.x * blockDim.x + threadIdx.x;
    if (i < n) p[i] = 0.0f;
}

__global__ void k_deg_count(const int* __restrict__ src, const int* __restrict__ dst,
                            float* __restrict__ deg_out, float* __restrict__ deg_in) {
    int i = blockIdx.x * blockDim.x + threadIdx.x;
    if (i < Rr * Ee) {
        int r = i / Ee;
        atomicAdd(deg_out + r * Nn + src[i], 1.0f);
        atomicAdd(deg_in  + r * Nn + dst[i], 1.0f);
    }
}

__global__ void k_deg_fin(float* __restrict__ p, int n) {
    int i = blockIdx.x * blockDim.x + threadIdx.x;
    if (i < n) p[i] = rsqrtf(fmaxf(p[i], 1.0f));
}

// f32 -> bf16 (optionally ReLU)
__global__ void k_cvt(const float* __restrict__ in, __bf16* __restrict__ out, int n, int relu) {
    int i = blockIdx.x * blockDim.x + threadIdx.x;
    if (i < n) {
        float v = in[i];
        if (relu) v = fmaxf(v, 0.0f);
        out[i] = (__bf16)v;
    }
}

// WT[r][n][k] = bf16(W[r][k][n])  : makes B-fragment loads contiguous in k
__global__ void k_wt(const float* __restrict__ W, __bf16* __restrict__ WT) {
    int i = blockIdx.x * blockDim.x + threadIdx.x;
    if (i < Rr * Dd * Dd) {
        int r = i / (Dd * Dd);
        int rem = i - r * Dd * Dd;
        int k = rem / Dd;
        int n = rem - k * Dd;
        WT[r * Dd * Dd + n * Dd + k] = (__bf16)W[i];
    }
}

// acc[n][d] = b1[0][d] + b1[1][d] + b1[2][d]  (sum of per-relation bias broadcasts)
__global__ void k_bias(const float* __restrict__ b, float* __restrict__ acc) {
    int i = blockIdx.x * blockDim.x + threadIdx.x;
    if (i < Nn * Dd) {
        int d = i & (Dd - 1);
        acc[i] = b[d] + b[Dd + d] + b[2 * Dd + d];
    }
}

// ---------------- WMMA GEMM: Cb[N,128] = A[N,128] @ W  (B given transposed: BT[n][k]) ----------

__global__ void k_gemm(const __bf16* __restrict__ A, const __bf16* __restrict__ BT,
                       __bf16* __restrict__ Cb) {
    const int lane = threadIdx.x & 31;
    const int wave = threadIdx.x >> 5;          // 8 waves -> 8 column tiles (full 128)
    const int row0 = blockIdx.x * 16;
    const int col0 = wave * 16;
    const int half = lane >> 4;
    const int idx  = lane & 15;

    const __bf16* arow = A  + (size_t)(row0 + idx) * Dd;              // A row, K contiguous
    const __bf16* brow = BT + (size_t)(col0 + idx) * Dd + half * 16;  // B col (as WT row)

    f32x8 c = {0.f, 0.f, 0.f, 0.f, 0.f, 0.f, 0.f, 0.f};
#pragma unroll
    for (int kb = 0; kb < 4; ++kb) {
        const int kbase = kb * 32;
        // A 16x32 bf16 fragment: lane=row, K = half*16 + e (contiguous, matches B)
        bf16x8 a_lo = *(const bf16x8*)(arow + kbase + half * 16);
        bf16x8 a_hi = *(const bf16x8*)(arow + kbase + half * 16 + 8);
        bf16x16 a = __builtin_shufflevector(a_lo, a_hi,
                                            0, 1, 2, 3, 4, 5, 6, 7,
                                            8, 9, 10, 11, 12, 13, 14, 15);
        // B 32x16 bf16 fragment: lane=col n, K = half*16 + e (contiguous 32B in WT row)
        bf16x16 b = *(const bf16x16*)(brow + kbase);
        c = __builtin_amdgcn_wmma_f32_16x16x32_bf16(false, a, false, b,
                                                    (short)0, c, false, false);
    }
    // D f32 16x16: VGPR v, lane l -> M = (l>>4)*8 + v, N = l&15
    __bf16* out = Cb + (size_t)(row0 + half * 8) * Dd + col0 + idx;
#pragma unroll
    for (int v = 0; v < 8; ++v) out[(size_t)v * Dd] = (__bf16)c[v];
}

// ---------------- edge scatter: acc[dst] += hw[src] * inv_out[src] * inv_in[dst] --------------

__global__ void k_edge(const __bf16* __restrict__ hw, const int* __restrict__ src,
                       const int* __restrict__ dst, const float* __restrict__ invo,
                       const float* __restrict__ invi, float* __restrict__ acc) {
    int i = blockIdx.x * blockDim.x + threadIdx.x;   // 32 threads per edge
    int e = i >> 5;
    if (e < Ee) {
        int ch = i & 31;
        int s = src[e];
        int d = dst[e];
        float sc = invo[s] * invi[d];
        bf16x4 hv = *(const bf16x4*)(hw + (size_t)s * Dd + ch * 4);
        float* ap = acc + (size_t)d * Dd + ch * 4;
        atomicAdd(ap + 0, (float)hv[0] * sc);
        atomicAdd(ap + 1, (float)hv[1] * sc);
        atomicAdd(ap + 2, (float)hv[2] * sc);
        atomicAdd(ap + 3, (float)hv[3] * sc);
    }
}

// ---------------- pooling + head ----------------

__global__ void k_pool_cnt(const int* __restrict__ gid, float* __restrict__ counts) {
    int i = blockIdx.x * blockDim.x + threadIdx.x;
    if (i < Nn) atomicAdd(counts + gid[i], 1.0f);
}

__global__ void k_pool_sum(const float* __restrict__ acc, const int* __restrict__ gid,
                           float* __restrict__ hg) {
    int i = blockIdx.x * blockDim.x + threadIdx.x;   // 32 threads per node
    int n = i >> 5;
    if (n < Nn) {
        int ch = i & 31;
        int g = gid[n];
        const float* ap = acc + (size_t)n * Dd + ch * 4;
        float* hp = hg + (size_t)g * Dd + ch * 4;
        atomicAdd(hp + 0, ap[0]);
        atomicAdd(hp + 1, ap[1]);
        atomicAdd(hp + 2, ap[2]);
        atomicAdd(hp + 3, ap[3]);
    }
}

// logits = (hg/cnt) @ Wc + bc ; softmax over axis=0 (over graphs, per class) as in reference
__global__ void k_head(const float* __restrict__ hg, const float* __restrict__ counts,
                       const float* __restrict__ Wc, const float* __restrict__ bc,
                       float* __restrict__ out) {
    __shared__ float lds[Bb * Cc];
    int t = threadIdx.x;                 // 640 threads
    int b = t / Cc;
    int c = t - b * Cc;
    float cnt = fmaxf(counts[b], 1.0f);
    float s = 0.0f;
    for (int d = 0; d < Dd; ++d) s += hg[b * Dd + d] * Wc[d * Cc + c];
    s = s / cnt + bc[c];
    lds[t] = s;
    __syncthreads();
    float m = -3.4e38f;
    for (int bb = 0; bb < Bb; ++bb) m = fmaxf(m, lds[bb * Cc + c]);
    float z = 0.0f;
    for (int bb = 0; bb < Bb; ++bb) z += expf(lds[bb * Cc + c] - m);
    out[t] = expf(s - m) / z;
}

// ---------------- launch ----------------

static inline size_t alignup(size_t x) { return (x + 255) & ~(size_t)255; }

extern "C" void kernel_launch(void* const* d_in, const int* in_sizes, int n_in,
                              void* d_out, int out_size, void* d_ws, size_t ws_size,
                              hipStream_t stream) {
    (void)in_sizes; (void)n_in; (void)out_size; (void)ws_size;

    const float* feature = (const float*)d_in[0];
    const int*   src     = (const int*)  d_in[1];   // [R,E]
    const int*   dst     = (const int*)  d_in[2];   // [R,E]
    const int*   gid     = (const int*)  d_in[3];   // [N]
    const float* W1      = (const float*)d_in[4];
    const float* b1      = (const float*)d_in[5];
    const float* W2      = (const float*)d_in[6];
    const float* b2      = (const float*)d_in[7];
    const float* Wc      = (const float*)d_in[8];
    const float* bc      = (const float*)d_in[9];
    float* out = (float*)d_out;

    // workspace carving (~105 MB)
    char* w = (char*)d_ws;
    size_t off = 0;
    __bf16* h_b  = (__bf16*)(w + off); off = alignup(off + (size_t)Nn * Dd * 2);
    __bf16* hw_b = (__bf16*)(w + off); off = alignup(off + (size_t)Nn * Dd * 2);
    float*  acc  = (float*) (w + off); off = alignup(off + (size_t)Nn * Dd * 4);
    __bf16* WT   = (__bf16*)(w + off); off = alignup(off + (size_t)Rr * Dd * Dd * 2);
    float*  invo = (float*) (w + off); off = alignup(off + (size_t)Rr * Nn * 4);
    float*  invi = (float*) (w + off); off = alignup(off + (size_t)Rr * Nn * 4);
    float*  hg   = (float*) (w + off); off = alignup(off + (size_t)Bb * Dd * 4 + Bb * 4);
    float*  counts = hg + Bb * Dd;     // contiguous with hg

    const int TB = 256;
    const int gRN   = (Rr * Nn + TB - 1) / TB;
    const int gRE   = (Rr * Ee + TB - 1) / TB;
    const int gND   = (Nn * Dd + TB - 1) / TB;
    const int gRDD  = (Rr * Dd * Dd + TB - 1) / TB;
    const int gEdge = (Ee * 32) / TB;          // 25000
    const int gGemm = Nn / 16;                 // 6250
    const int gN    = (Nn + TB - 1) / TB;
    const int gN32  = (Nn * 32 + TB - 1) / TB;

    // degrees (shared by both layers)
    k_zero<<<gRN, TB, 0, stream>>>(invo, Rr * Nn);
    k_zero<<<gRN, TB, 0, stream>>>(invi, Rr * Nn);
    k_deg_count<<<gRE, TB, 0, stream>>>(src, dst, invo, invi);
    k_deg_fin<<<gRN, TB, 0, stream>>>(invo, Rr * Nn);
    k_deg_fin<<<gRN, TB, 0, stream>>>(invi, Rr * Nn);

    // ---- layer 1 ----
    k_cvt<<<gND, TB, 0, stream>>>(feature, h_b, Nn * Dd, 0);
    k_wt<<<gRDD, TB, 0, stream>>>(W1, WT);
    k_bias<<<gND, TB, 0, stream>>>(b1, acc);
    for (int r = 0; r < Rr; ++r) {
        k_gemm<<<gGemm, TB, 0, stream>>>(h_b, WT + (size_t)r * Dd * Dd, hw_b);
        k_edge<<<gEdge, TB, 0, stream>>>(hw_b, src + (size_t)r * Ee, dst + (size_t)r * Ee,
                                         invo + (size_t)r * Nn, invi + (size_t)r * Nn, acc);
    }

    // ---- layer 2 (ReLU folded into conversion) ----
    k_cvt<<<gND, TB, 0, stream>>>(acc, h_b, Nn * Dd, 1);
    k_wt<<<gRDD, TB, 0, stream>>>(W2, WT);
    k_bias<<<gND, TB, 0, stream>>>(b2, acc);
    for (int r = 0; r < Rr; ++r) {
        k_gemm<<<gGemm, TB, 0, stream>>>(h_b, WT + (size_t)r * Dd * Dd, hw_b);
        k_edge<<<gEdge, TB, 0, stream>>>(hw_b, src + (size_t)r * Ee, dst + (size_t)r * Ee,
                                         invo + (size_t)r * Nn, invi + (size_t)r * Nn, acc);
    }

    // ---- mean pool + head ----
    k_zero<<<(Bb * Dd + Bb + TB - 1) / TB, TB, 0, stream>>>(hg, Bb * Dd + Bb);
    k_pool_cnt<<<gN, TB, 0, stream>>>(gid, counts);
    k_pool_sum<<<gN32, TB, 0, stream>>>(acc, gid, hg);
    k_head<<<1, Bb * Cc, 0, stream>>>(hg, counts, Wc, bc, out);
}
